// UniRelModel_71837622993443
// MI455X (gfx1250) — matrix operand order repaired and checked by reading, not compiled, not run on hardware
//
#include <hip/hip_runtime.h>

#define H      768
#define FF     3072
#define NLAYER 12
#define NHEAD  12
#define DHEAD  64
#define SEQ    512
#define BATCH  4
#define TOK    (BATCH*SEQ)
#define NE     12
#define NR     13

typedef __attribute__((ext_vector_type(16))) __bf16 v16bf;
typedef __attribute__((ext_vector_type(8)))  __bf16 v8bf;
typedef __attribute__((ext_vector_type(8)))  float  v8f;

#define BF16CAT(lo,hi) __builtin_shufflevector((lo),(hi),0,1,2,3,4,5,6,7,8,9,10,11,12,13,14,15)
#define WMMA_BF16(a,b,c) __builtin_amdgcn_wmma_f32_16x16x32_bf16(false,(a),false,(b),(short)0,(c),false,false)

#ifndef __has_builtin
#define __has_builtin(x) 0
#endif
#if defined(__gfx1250__) && __has_builtin(__builtin_amdgcn_tensor_load_to_lds) && \
    __has_builtin(__builtin_amdgcn_s_wait_tensorcnt)
#define HAVE_TDM 1
#else
#define HAVE_TDM 0
#endif

__device__ __forceinline__ unsigned short f2bf(float f) {
  unsigned int u = __float_as_uint(f);
  u += 0x7fffu + ((u >> 16) & 1u);           // round-to-nearest-even
  return (unsigned short)(u >> 16);
}

// Pack two fp32 -> packed bf16x2 entirely in 32-bit integer ALU
// (avoids 16-bit subregister mov chains).
__device__ __forceinline__ unsigned pack2_bf16(float a, float b) {
  unsigned ua = __float_as_uint(a);
  unsigned ub = __float_as_uint(b);
  ua += 0x7fffu + ((ua >> 16) & 1u);
  ub += 0x7fffu + ((ub >> 16) & 1u);
  return (ua >> 16) | (ub & 0xffff0000u);
}

#if HAVE_TDM
typedef __attribute__((ext_vector_type(4))) unsigned int u32x4;
typedef __attribute__((ext_vector_type(8))) int          i32x8;
typedef __attribute__((ext_vector_type(4))) int          i32x4;

// 2D tile load via Tensor Data Mover. data_size = 2 bytes (bf16).
// LDS padding: padI -> pad every 2^(padI+1) DWORDs, padA -> (padA+1) DWORDs pad.
__device__ __forceinline__ void tdm_load_2d(const void* g, unsigned lds_off,
                                            unsigned tile0, unsigned tile1,
                                            unsigned tdim0, unsigned tdim1,
                                            unsigned long long stride0,
                                            unsigned padI, unsigned padA)
{
  const unsigned long long ga = (unsigned long long)(size_t)g;
  u32x4 g0;
  g0.x = 1u;                                              // count=1, user mode, no gather
  g0.y = lds_off;                                         // lds_addr [63:32]
  g0.z = (unsigned)ga;                                    // global_addr [95:64]
  g0.w = (unsigned)((ga >> 32) & 0x01FFFFFFu) | (2u << 30); // addr[56:32] | type=2
  i32x8 g1;
  g1[0] = (int)((1u << 16) | (1u << 20) | (padI << 22) | (padA << 25)); // 2B, pad_enable
  g1[1] = (int)((tdim0 & 0xFFFFu) << 16);                 // tensor_dim0[15:0] @ bits 63:48
  g1[2] = (int)((tdim0 >> 16) | ((tdim1 & 0xFFFFu) << 16));
  g1[3] = (int)((tdim1 >> 16) | (tile0 << 16));           // tile_dim0 @ bits 127:112
  g1[4] = (int)(tile1 & 0xFFFFu);                         // tile_dim1; tile_dim2 = 0
  g1[5] = (int)(stride0 & 0xFFFFFFFFull);                 // tensor_dim0_stride lo
  g1[6] = (int)((stride0 >> 32) & 0xFFFFull);             // stride hi; dim1_stride = 0
  g1[7] = 0;
  i32x4 gz4 = {};
#if __clang_major__ >= 23
  i32x8 gz8 = {};
  __builtin_amdgcn_tensor_load_to_lds(g0, g1, gz4, gz4, gz8, 0);
#else
  __builtin_amdgcn_tensor_load_to_lds(g0, g1, gz4, gz4, 0);
#endif
}
#endif

// =====================================================================
// GEMM: out[M][N] = bf16 A[M][K] * f32 W[K][N] + bias[N]
// epi: 0 = fp32 out, 1 = bf16 out, 2 = bf16 out with exact GELU
// Tiles: BM=128 BN=128 BK=32; 8 waves in 2(m)x4(n); each wave 4x2 WMMAs.
// A tile via TDM (double buffered) when available; W tile converted
// fp32->bf16 and stored transposed with packed b128 LDS stores.
// =====================================================================
__global__ __launch_bounds__(256)
void gemm_wmma(const unsigned short* __restrict__ A,
               const float* __restrict__ W,
               const float* __restrict__ bias,
               void* __restrict__ out,
               int M, int K, int N, int epi)
{
  constexpr int BM = 128, BN = 128, BK = 32, BKP = 40;
  __shared__ alignas(16) unsigned short As[2][BM][BKP];
  __shared__ alignas(16) unsigned short Bs[BN][BKP];     // transposed: [n][k]

  const int tid  = threadIdx.x;
  const int wave = tid >> 5, lane = tid & 31;
  const int llo  = lane & 15, lhi = lane >> 4;
  const int wm   = wave >> 2, wn  = wave & 3;
  const int m0   = blockIdx.y * BM, n0 = blockIdx.x * BN;

  const v8f vzero = {};
  v8f acc[4][2];
#pragma unroll
  for (int i = 0; i < 4; i++)
#pragma unroll
    for (int j = 0; j < 2; j++) acc[i][j] = vzero;

  // ---- W tile staging: thread owns column n, 16 consecutive k ----
  const int wcol = tid & 127, wkh = (tid >> 7) * 16;
  float wReg[16];
  auto loadW = [&](int kt) {
#pragma unroll
    for (int i = 0; i < 16; i++)
      wReg[i] = W[(size_t)(kt + wkh + i) * N + n0 + wcol];
    if (kt + BK < K)
      __builtin_prefetch(W + (size_t)(kt + BK + wkh) * N + n0 + wcol, 0, 3);
  };
  auto storeW = [&]() {
    unsigned p[8];
#pragma unroll
    for (int i = 0; i < 8; i++)
      p[i] = pack2_bf16(wReg[2*i], wReg[2*i + 1]);
    *(uint4*)&Bs[wcol][wkh]     = make_uint4(p[0], p[1], p[2], p[3]);
    *(uint4*)&Bs[wcol][wkh + 8] = make_uint4(p[4], p[5], p[6], p[7]);
  };

#if HAVE_TDM
  auto tdmA = [&](int buf, int kt) {
    if (wave == 0)
      tdm_load_2d(A + (size_t)m0 * K + kt,
                  (unsigned)(size_t)&As[buf][0][0],
                  /*tile0*/BK, /*tile1*/BM, /*tdim0*/BK, /*tdim1*/BM,
                  /*stride0*/(unsigned long long)K,
                  /*padI: 16 DW (=64B row)*/3, /*padA: 4 DW (=16B)*/3);
  };
#else
  const int arow = tid >> 1, acol = (tid & 1) * 16;
  uint4 aReg0, aReg1;
  auto loadA = [&](int kt) {
    const uint4* ap = (const uint4*)(A + (size_t)(m0 + arow) * K + kt + acol);
    aReg0 = ap[0]; aReg1 = ap[1];
  };
  auto storeA = [&]() {
    *(uint4*)&As[0][arow][acol]     = aReg0;
    *(uint4*)&As[0][arow][acol + 8] = aReg1;
  };
#endif

  loadW(0);
#if HAVE_TDM
  tdmA(0, 0);
#else
  loadA(0);
#endif

  for (int kt = 0, it = 0; kt < K; kt += BK, ++it) {
#if HAVE_TDM
    const int buf = it & 1;
#else
    const int buf = 0;
#endif
    __syncthreads();
    storeW();
#if HAVE_TDM
    if (wave == 0) {
      if (kt + BK < K) { tdmA(buf ^ 1, kt + BK); __builtin_amdgcn_s_wait_tensorcnt(1); }
      else             { __builtin_amdgcn_s_wait_tensorcnt(0); }
    }
#else
    storeA();
#endif
    __syncthreads();
    if (kt + BK < K) {
      loadW(kt + BK);
#if !HAVE_TDM
      loadA(kt + BK);
#endif
    }

    v16bf afr[4], bfr[2];
#pragma unroll
    for (int mt = 0; mt < 4; mt++) {
      const unsigned short* p = &As[buf][wm * 64 + mt * 16 + llo][lhi * 8];
      v8bf lo = *(const v8bf*)p;
      v8bf hi = *(const v8bf*)(p + 16);
      afr[mt] = BF16CAT(lo, hi);
    }
#pragma unroll
    for (int nt = 0; nt < 2; nt++) {
      const unsigned short* p = &Bs[wn * 32 + nt * 16 + llo][lhi * 8];
      v8bf lo = *(const v8bf*)p;
      v8bf hi = *(const v8bf*)(p + 16);
      bfr[nt] = BF16CAT(lo, hi);
    }
#pragma unroll
    for (int mt = 0; mt < 4; mt++)
#pragma unroll
      for (int nt = 0; nt < 2; nt++)
        acc[mt][nt] = WMMA_BF16(afr[mt], bfr[nt], acc[mt][nt]);
  }

  // epilogue (C layout: lane = column n, VGPR r = row r + 8*lhi)
#pragma unroll
  for (int nt = 0; nt < 2; nt++) {
    const int ng = n0 + wn * 32 + nt * 16 + llo;
    const float bv = bias[ng];
#pragma unroll
    for (int mt = 0; mt < 4; mt++) {
      const int mg0 = m0 + wm * 64 + mt * 16 + lhi * 8;
#pragma unroll
      for (int r = 0; r < 8; r++) {
        float v = acc[mt][nt][r] + bv;
        size_t idx = (size_t)(mg0 + r) * N + ng;
        if (epi == 0) {
          ((float*)out)[idx] = v;
        } else {
          if (epi == 2) v = 0.5f * v * (1.0f + erff(v * 0.70710678118654752f));
          ((unsigned short*)out)[idx] = f2bf(v);
        }
      }
    }
  }
}

// =====================================================================
// Flash attention: block = (128 q-rows, head, batch); 8 waves x 16 rows.
// Online softmax entirely in registers (C-layout rows live per-VGPR).
// K tile staged by TDM when available; V tile transposed with packed stores.
// =====================================================================
__global__ __launch_bounds__(256)
void flash_attn(const unsigned short* __restrict__ Qm,
                const unsigned short* __restrict__ Km,
                const unsigned short* __restrict__ Vm,
                const int* __restrict__ mask,
                unsigned short* __restrict__ ctx)
{
  __shared__ alignas(16) unsigned short Ks[64][72];
  __shared__ alignas(16) unsigned short VsT[64][72];       // transposed: [d][j]
  __shared__ alignas(16) unsigned short Ps[8][16][72];     // per-wave P tiles

  const int tid  = threadIdx.x;
  const int wave = tid >> 5, lane = tid & 31;
  const int llo  = lane & 15, lhi = lane >> 4;
  const int i0   = blockIdx.x * 128;
  const int h    = blockIdx.y, b = blockIdx.z;
  const size_t rowBase = (size_t)b * SEQ;

  const v8f vzero = {};

  // Q fragments for this wave's 16 rows (A-layout, two 32-K steps over DH=64)
  v16bf qf[2];
  {
    const unsigned short* qp = Qm + (rowBase + i0 + wave * 16 + llo) * H + h * DHEAD;
#pragma unroll
    for (int ks = 0; ks < 2; ks++) {
      const int kb0 = ks * 32 + lhi * 8;
      v8bf lo = *(const v8bf*)(qp + kb0);
      v8bf hi = *(const v8bf*)(qp + kb0 + 16);
      qf[ks] = BF16CAT(lo, hi);
    }
  }

  float mrow[8], lrow[8];
  v8f o[4];
#pragma unroll
  for (int r = 0; r < 8; r++) { mrow[r] = -1e30f; lrow[r] = 0.f; }
#pragma unroll
  for (int nt = 0; nt < 4; nt++) o[nt] = vzero;

  for (int j0 = 0; j0 < SEQ; j0 += 64) {
    __syncthreads();
#if HAVE_TDM
    if (wave == 0)
      tdm_load_2d(Km + (rowBase + j0) * H + h * DHEAD,
                  (unsigned)(size_t)&Ks[0][0],
                  /*tile0*/64, /*tile1*/64, /*tdim0*/64, /*tdim1*/64,
                  /*stride0*/(unsigned long long)H,
                  /*padI: 32 DW (=128B row)*/4, /*padA: 4 DW*/3);
#else
    { // stage K tile [j][d]
      const int srow = tid >> 2, sch = tid & 3;
      const uint4* src = (const uint4*)(Km + (rowBase + j0 + srow) * H + h * DHEAD + sch * 16);
      uint4 x0 = src[0], x1 = src[1];
      *(uint4*)&Ks[srow][sch * 16]     = x0;
      *(uint4*)&Ks[srow][sch * 16 + 8] = x1;
    }
#endif
    { // stage V tile transposed -> VsT[d][j], packed b128 stores
      const int d = tid & 63, jh = (tid >> 6) * 16;
      const unsigned short* vp = Vm + (rowBase + j0 + jh) * H + h * DHEAD + d;
      unsigned p[8];
#pragma unroll
      for (int i = 0; i < 8; i++) {
        unsigned lo = vp[(size_t)(2*i) * H];
        unsigned hi = vp[(size_t)(2*i + 1) * H];
        p[i] = lo | (hi << 16);
      }
      *(uint4*)&VsT[d][jh]     = make_uint4(p[0], p[1], p[2], p[3]);
      *(uint4*)&VsT[d][jh + 8] = make_uint4(p[4], p[5], p[6], p[7]);
    }
#if HAVE_TDM
    if (wave == 0) __builtin_amdgcn_s_wait_tensorcnt(0);
#endif
    __syncthreads();

    // S = Q K^T  (B-fragment: lane = column j, d in two runs of 8 at 8*lhi and 16 + 8*lhi)
    v8f sc4[4];
#pragma unroll
    for (int jt = 0; jt < 4; jt++) sc4[jt] = vzero;
#pragma unroll
    for (int ks = 0; ks < 2; ks++)
#pragma unroll
      for (int jt = 0; jt < 4; jt++) {
        const unsigned short* p = &Ks[jt * 16 + llo][ks * 32 + lhi * 8];
        v8bf lo = *(const v8bf*)p;
        v8bf hi = *(const v8bf*)(p + 16);
        v16bf kf = BF16CAT(lo, hi);
        sc4[jt] = WMMA_BF16(qf[ks], kf, sc4[jt]);
      }
#pragma unroll
    for (int jt = 0; jt < 4; jt++) {
      const int j = j0 + jt * 16 + llo;
      const float mb = (1.f - (float)mask[rowBase + j]) * -10000.f;
#pragma unroll
      for (int r = 0; r < 8; r++) sc4[jt][r] = sc4[jt][r] * 0.125f + mb;
    }

    // online softmax: row = r + 8*lhi, reduce across the 16-lane half
    float mnew[8], scal[8], rs[8];
#pragma unroll
    for (int r = 0; r < 8; r++) {
      float mx = fmaxf(fmaxf(sc4[0][r], sc4[1][r]), fmaxf(sc4[2][r], sc4[3][r]));
#pragma unroll
      for (int off = 1; off < 16; off <<= 1) mx = fmaxf(mx, __shfl_xor(mx, off, 32));
      mnew[r] = fmaxf(mrow[r], mx);
      scal[r] = __expf(mrow[r] - mnew[r]);
      rs[r]   = 0.f;
    }
#pragma unroll
    for (int jt = 0; jt < 4; jt++)
#pragma unroll
      for (int r = 0; r < 8; r++) {
        float p = __expf(sc4[jt][r] - mnew[r]);
        rs[r] += p;
        Ps[wave][r + 8 * lhi][jt * 16 + llo] = f2bf(p);
      }
#pragma unroll
    for (int r = 0; r < 8; r++) {
      float t = rs[r];
#pragma unroll
      for (int off = 1; off < 16; off <<= 1) t += __shfl_xor(t, off, 32);
      lrow[r] = lrow[r] * scal[r] + t;
      mrow[r] = mnew[r];
#pragma unroll
      for (int nt = 0; nt < 4; nt++) o[nt][r] *= scal[r];
    }
    __syncthreads();

    // O += P V   (A from Ps, B from VsT)
#pragma unroll
    for (int ks = 0; ks < 2; ks++) {
      const unsigned short* pp = &Ps[wave][llo][ks * 32 + lhi * 8];
      v8bf plo = *(const v8bf*)pp;
      v8bf phi = *(const v8bf*)(pp + 16);
      v16bf pf = BF16CAT(plo, phi);
#pragma unroll
      for (int nt = 0; nt < 4; nt++) {
        const unsigned short* vp = &VsT[nt * 16 + llo][ks * 32 + lhi * 8];
        v8bf vlo = *(const v8bf*)vp;
        v8bf vhi = *(const v8bf*)(vp + 16);
        v16bf vf = BF16CAT(vlo, vhi);
        o[nt] = WMMA_BF16(pf, vf, o[nt]);
      }
    }
  }

#pragma unroll
  for (int nt = 0; nt < 4; nt++) {
    const int d = h * DHEAD + nt * 16 + llo;
#pragma unroll
    for (int r = 0; r < 8; r++) {
      const int irow = i0 + wave * 16 + r + 8 * lhi;
      float v = o[nt][r] / lrow[r];
      ctx[(rowBase + irow) * H + d] = f2bf(v);
    }
  }
}

// =====================================================================
// LayerNorm helpers / elementwise kernels
// =====================================================================
__device__ __forceinline__ float blockReduceSum(float v, float* red) {
#pragma unroll
  for (int off = 16; off > 0; off >>= 1) v += __shfl_xor(v, off, 32);
  const int wv = threadIdx.x >> 5, ln = threadIdx.x & 31;
  __syncthreads();
  if (ln == 0) red[wv] = v;
  __syncthreads();
  float s = 0.f;
#pragma unroll
  for (int i = 0; i < 8; i++) s += red[i];
  return s;
}

__global__ __launch_bounds__(256)
void embed_ln(const int* __restrict__ ids, const float* __restrict__ ew,
              const float* __restrict__ ep, const float* __restrict__ et,
              const float* __restrict__ g, const float* __restrict__ bta,
              float* __restrict__ xo, unsigned short* __restrict__ xbo)
{
  __shared__ float red[8];
  const int t = blockIdx.x, tid = threadIdx.x;
  const int s_pos = t & (SEQ - 1);
  const int id = ids[t];
  const size_t base = (size_t)t * H;
  float v[3]; float s = 0.f;
#pragma unroll
  for (int i = 0; i < 3; i++) {
    int c = tid + 256 * i;
    v[i] = ew[(size_t)id * H + c] + ep[(size_t)s_pos * H + c] + et[c];
    s += v[i];
  }
  float mean = blockReduceSum(s, red) * (1.f / H);
  float s2 = 0.f;
#pragma unroll
  for (int i = 0; i < 3; i++) { float d = v[i] - mean; s2 += d * d; }
  float rstd = rsqrtf(blockReduceSum(s2, red) * (1.f / H) + 1e-12f);
#pragma unroll
  for (int i = 0; i < 3; i++) {
    int c = tid + 256 * i;
    float y = (v[i] - mean) * rstd * g[c] + bta[c];
    xo[base + c] = y;
    xbo[base + c] = f2bf(y);
  }
}

__global__ __launch_bounds__(256)
void add_ln(const float* __restrict__ x, const float* __restrict__ d,
            const float* __restrict__ g, const float* __restrict__ bta,
            float* __restrict__ xo, unsigned short* __restrict__ xbo)
{
  __shared__ float red[8];
  const int row = blockIdx.x, tid = threadIdx.x;
  const size_t base = (size_t)row * H;
  float v[3]; float s = 0.f;
#pragma unroll
  for (int i = 0; i < 3; i++) {
    int c = tid + 256 * i;
    v[i] = x[base + c] + d[base + c];
    s += v[i];
  }
  float mean = blockReduceSum(s, red) * (1.f / H);
  float s2 = 0.f;
#pragma unroll
  for (int i = 0; i < 3; i++) { float t = v[i] - mean; s2 += t * t; }
  float rstd = rsqrtf(blockReduceSum(s2, red) * (1.f / H) + 1e-12f);
#pragma unroll
  for (int i = 0; i < 3; i++) {
    int c = tid + 256 * i;
    float y = (v[i] - mean) * rstd * g[c] + bta[c];
    xo[base + c] = y;
    xbo[base + c] = f2bf(y);
  }
}

// =====================================================================
// Output heads: entity logits + pi/pj projections, then broadcast-add.
// =====================================================================
__global__ __launch_bounds__(256)
void heads(const float* __restrict__ x,
           const float* __restrict__ Went, const float* __restrict__ bent,
           const float* __restrict__ Wrel, const float* __restrict__ brel,
           float* __restrict__ ent, float* __restrict__ pi, float* __restrict__ pj)
{
  __shared__ float xs[H];
  const int t = blockIdx.x, tid = threadIdx.x;
#pragma unroll
  for (int i = 0; i < 3; i++) xs[tid + 256 * i] = x[(size_t)t * H + tid + 256 * i];
  __syncthreads();
  const int wave = tid >> 5, lane = tid & 31;
  for (int o = wave; o < NE + 2 * NR; o += 8) {
    float s = 0.f;
    if (o < NE) {
      for (int c = lane; c < H; c += 32) s += xs[c] * Went[(size_t)c * NE + o];
    } else if (o < NE + NR) {
      const int r = o - NE;
      for (int c = lane; c < H; c += 32) s += xs[c] * Wrel[(size_t)c * NR + r];
    } else {
      const int r = o - NE - NR;
      for (int c = lane; c < H; c += 32) s += xs[c] * Wrel[(size_t)(H + c) * NR + r];
    }
#pragma unroll
    for (int off = 16; off > 0; off >>= 1) s += __shfl_xor(s, off, 32);
    if (lane == 0) {
      if (o < NE)            ent[(size_t)t * NE + o] = s + bent[o];
      else if (o < NE + NR)  pi[t * NR + (o - NE)] = s;
      else                   pj[t * NR + (o - NE - NR)] = s + brel[o - NE - NR];
    }
  }
}

__global__ __launch_bounds__(256)
void rel_bcast(const float* __restrict__ pi, const float* __restrict__ pj,
               float* __restrict__ out)
{
  size_t idx = (size_t)blockIdx.x * 256 + threadIdx.x;
  const size_t total = (size_t)BATCH * SEQ * SEQ * NR;
  if (idx >= total) return;
  const int r = (int)(idx % NR);
  size_t q = idx / NR;
  const int j = (int)(q % SEQ); q /= SEQ;
  const int i = (int)(q % SEQ);
  const int b = (int)(q / SEQ);
  out[idx] = pi[((size_t)b * SEQ + i) * NR + r] + pj[((size_t)b * SEQ + j) * NR + r];
}

// =====================================================================
// Driver
// =====================================================================
extern "C" void kernel_launch(void* const* d_in, const int* in_sizes, int n_in,
                              void* d_out, int out_size, void* d_ws, size_t ws_size,
                              hipStream_t stream)
{
  (void)in_sizes; (void)n_in; (void)out_size; (void)ws_size;
  const int*   ids  = (const int*)d_in[0];
  const int*   msk  = (const int*)d_in[1];
  const float* ew   = (const float*)d_in[2];
  const float* ep   = (const float*)d_in[3];
  const float* et   = (const float*)d_in[4];
  const float* elg  = (const float*)d_in[5];
  const float* elb  = (const float*)d_in[6];
  const float* Wq   = (const float*)d_in[7];
  const float* bq   = (const float*)d_in[8];
  const float* Wk   = (const float*)d_in[9];
  const float* bk   = (const float*)d_in[10];
  const float* Wv   = (const float*)d_in[11];
  const float* bv   = (const float*)d_in[12];
  const float* Wo   = (const float*)d_in[13];
  const float* bo   = (const float*)d_in[14];
  const float* ln1g = (const float*)d_in[15];
  const float* ln1b = (const float*)d_in[16];
  const float* W1   = (const float*)d_in[17];
  const float* b1   = (const float*)d_in[18];
  const float* W2   = (const float*)d_in[19];
  const float* b2   = (const float*)d_in[20];
  const float* ln2g = (const float*)d_in[21];
  const float* ln2b = (const float*)d_in[22];
  const float* Went = (const float*)d_in[23];
  const float* bent = (const float*)d_in[24];
  const float* Wrel = (const float*)d_in[25];
  const float* brel = (const float*)d_in[26];

  // workspace carve-out (~41.2 MB)
  char* ws = (char*)d_ws;
  size_t off = 0;
  auto carve = [&](size_t bytes) -> void* {
    void* p = ws + off;
    off = (off + bytes + 255) & ~(size_t)255;
    return p;
  };
  float*          xw   = (float*)carve((size_t)TOK * H * 4);
  float*          tmp  = (float*)carve((size_t)TOK * H * 4);
  unsigned short* xb   = (unsigned short*)carve((size_t)TOK * H * 2);
  unsigned short* qb   = (unsigned short*)carve((size_t)TOK * H * 2);
  unsigned short* kb   = (unsigned short*)carve((size_t)TOK * H * 2);
  unsigned short* vb   = (unsigned short*)carve((size_t)TOK * H * 2);
  unsigned short* ctxb = (unsigned short*)carve((size_t)TOK * H * 2);
  unsigned short* ffh  = (unsigned short*)carve((size_t)TOK * FF * 2);
  float*          pi   = (float*)carve((size_t)TOK * NR * 4);
  float*          pj   = (float*)carve((size_t)TOK * NR * 4);

  embed_ln<<<TOK, 256, 0, stream>>>(ids, ew, ep, et, elg, elb, xw, xb);

  const dim3 gH(H / 128, TOK / 128);    // (6,16)
  const dim3 gF(FF / 128, TOK / 128);   // (24,16)
  const dim3 gAtt(SEQ / 128, NHEAD, BATCH);

  for (int l = 0; l < NLAYER; l++) {
    const size_t oHH = (size_t)l * H * H;
    gemm_wmma<<<gH, 256, 0, stream>>>(xb, Wq + oHH, bq + (size_t)l * H, qb, TOK, H, H, 1);
    gemm_wmma<<<gH, 256, 0, stream>>>(xb, Wk + oHH, bk + (size_t)l * H, kb, TOK, H, H, 1);
    gemm_wmma<<<gH, 256, 0, stream>>>(xb, Wv + oHH, bv + (size_t)l * H, vb, TOK, H, H, 1);
    flash_attn<<<gAtt, 256, 0, stream>>>(qb, kb, vb, msk, ctxb);
    gemm_wmma<<<gH, 256, 0, stream>>>(ctxb, Wo + oHH, bo + (size_t)l * H, tmp, TOK, H, H, 0);
    add_ln<<<TOK, 256, 0, stream>>>(xw, tmp, ln1g + (size_t)l * H, ln1b + (size_t)l * H, xw, xb);
    gemm_wmma<<<gF, 256, 0, stream>>>(xb, W1 + (size_t)l * H * FF, b1 + (size_t)l * FF, ffh, TOK, H, FF, 2);
    gemm_wmma<<<gH, 256, 0, stream>>>(ffh, W2 + (size_t)l * FF * H, b2 + (size_t)l * H, tmp, TOK, FF, H, 0);
    add_ln<<<TOK, 256, 0, stream>>>(xw, tmp, ln2g + (size_t)l * H, ln2b + (size_t)l * H, xw, xb);
  }

  float* ent_out = (float*)d_out;
  float* rel_out = (float*)d_out + (size_t)TOK * NE;
  heads<<<TOK, 256, 0, stream>>>(xw, Went, bent, Wrel, brel, ent_out, pi, pj);

  const size_t relTotal = (size_t)BATCH * SEQ * SEQ * NR;
  rel_bcast<<<(unsigned)((relTotal + 255) / 256), 256, 0, stream>>>(pi, pj, rel_out);
}
